// RelativePositionEncoding_33818572489082
// MI455X (gfx1250) — matrix operand order repaired and checked by reading, not compiled, not run on hardware
//
#include <hip/hip_runtime.h>
#include <hip/hip_bf16.h>

#define BB 8
#define SS 1024
#define HH 128

typedef __attribute__((ext_vector_type(16))) __bf16        v16bf;
typedef __attribute__((ext_vector_type(8)))  float         v8f;
typedef __attribute__((ext_vector_type(4)))  unsigned int  v4u;
typedef __attribute__((ext_vector_type(4)))  int           v4i;
typedef __attribute__((ext_vector_type(8)))  int           v8i;

union FragBF16 {
  v16bf v;
  uint4 q[2];
};

union Pack8 {
  uint4 q;
  unsigned short s[8];
};

// ---------------------------------------------------------------------------
// Kernel 0: convert x (fp32) -> bf16 scratch for the WMMA phase.
// ---------------------------------------------------------------------------
__global__ __launch_bounds__(256) void cvt_x_bf16_kernel(
    const float* __restrict__ x, __bf16* __restrict__ xb, int n) {
  int idx = blockIdx.x * 256 + threadIdx.x;
  if (idx < n) xb[idx] = (__bf16)x[idx];
}

// ---------------------------------------------------------------------------
// Kernel 1: scores[b,i,j] = sum_h x[b,j,h]*rel[i,j,h]; softmax over j;
// write bf16 weights. One block per i, 8 waves. Streams rel once (512 MB
// total -> 23.3 TB/s roofline); global_prefetch_b8 runs ahead of the stream.
// ---------------------------------------------------------------------------
__global__ __launch_bounds__(256) void scores_softmax_kernel(
    const float* __restrict__ x,    // [B,S,H]
    const float* __restrict__ rel,  // [S,S,H]
    __bf16* __restrict__ w_out) {   // [B,S,S]
  __shared__ float sc[BB][SS];      // 32 KB
  __shared__ float mred[BB];
  __shared__ float rsum[BB];

  const int i    = blockIdx.x;
  const int tid  = threadIdx.x;
  const int lane = tid & 31;
  const int wv   = tid >> 5;

  // ---- scores: wave wv handles j = wv, wv+8, ... (128 j's per wave) ----
  for (int j = wv; j < SS; j += 8) {
    // prefetch the rel row we will need 8 wave-iterations ahead
    if (j + 64 < SS)
      __builtin_prefetch(rel + ((size_t)i * SS + j + 64) * HH + lane * 4, 0, 0);

    const float4 r4 =
        *(const float4*)(rel + ((size_t)i * SS + j) * HH + lane * 4);
    float part[BB];
#pragma unroll
    for (int b = 0; b < BB; ++b) {
      const float4 x4 =
          *(const float4*)(x + ((size_t)b * SS + j) * HH + lane * 4);
      part[b] = r4.x * x4.x + r4.y * x4.y + r4.z * x4.z + r4.w * x4.w;
    }
#pragma unroll
    for (int b = 0; b < BB; ++b) {
      float v = part[b];
#pragma unroll
      for (int off = 16; off > 0; off >>= 1) v += __shfl_xor(v, off, 32);
      part[b] = v;
    }
    if (lane < BB) sc[lane][j] = part[lane];  // lane b writes batch b
  }
  __syncthreads();

  // ---- softmax reductions: wave wv owns batch b = wv ----
  {
    const int b = wv;
    float m = -3.4e38f;
    for (int j = lane; j < SS; j += 32) m = fmaxf(m, sc[b][j]);
#pragma unroll
    for (int off = 16; off > 0; off >>= 1)
      m = fmaxf(m, __shfl_xor(m, off, 32));
    float s = 0.0f;
    for (int j = lane; j < SS; j += 32) s += __expf(sc[b][j] - m);
#pragma unroll
    for (int off = 16; off > 0; off >>= 1) s += __shfl_xor(s, off, 32);
    if (lane == 0) {
      mred[b] = m;
      rsum[b] = 1.0f / s;
    }
  }
  __syncthreads();

  // ---- coalesced normalized bf16 write ----
  for (int idx = tid; idx < BB * SS; idx += 256) {
    const int b = idx >> 10;
    const int j = idx & (SS - 1);
    const float v = __expf(sc[b][j] - mred[b]) * rsum[b];
    w_out[((size_t)b * SS + i) * SS + j] = (__bf16)v;
  }
}

// ---------------------------------------------------------------------------
// Kernel 2: out[b,i,h] = sum_j w[b,i,j]*x[b,j,h].
// Grid (S/32, B), 256 threads = 8 waves; wave wv owns h-slice [wv*16,+16).
// Per K-step (32 j):
//   - TDM (tensor_load_to_lds) stages the 32x32 bf16 A tile of w, issued by
//     wave 0, completion via s_wait_tensorcnt.
//   - Cooperative transposed staging of the 32x128 x tile -> lxT[h][j]:
//     each thread owns 2 adjacent j-rows x 8 h, packs row pairs into uint
//     and issues 8 static-offset ds_store_b32 (conflict-free: consecutive
//     lanes -> consecutive banks).
//   - Fragments are pure ds_load_b128: A = 2, B = 2 per lane; 2 WMMAs/wave
//     (i-rows 0-15 and 16-31) share one B fragment.
// ---------------------------------------------------------------------------
__global__ __launch_bounds__(256) void wmma_av_kernel(
    const __bf16* __restrict__ w,   // [B,S,S]
    const __bf16* __restrict__ xb,  // [B,S,H]
    float* __restrict__ out) {      // [B,S,H]
  __shared__ __bf16 lw[32][32];     // A tile (TDM target), 2 KB
  __shared__ __bf16 lxT[HH][40];    // x tile transposed [h][j], 10 KB

  const int b    = blockIdx.y;
  const int i0   = blockIdx.x * 32;
  const int tid  = threadIdx.x;
  const int lane = tid & 31;
  const int wv   = tid >> 5;
  const int h0   = wv * 16;
  const int col  = lane & 15;  // A row (M) / B col (N) / D col (N)
  const int hi   = lane >> 4;  // K-half select

  v8f acc0 = {};
  v8f acc1 = {};

  const unsigned lds_lw = (unsigned)(size_t)(&lw[0][0]);

  for (int kt = 0; kt < SS / 32; ++kt) {
    const int j0 = kt * 32;

    // ---- TDM: async A-tile load (one issue, wave 0) ----
    if (wv == 0) {
      const unsigned long long ga =
          (unsigned long long)(size_t)(w + ((size_t)b * SS + i0) * SS + j0);
      v4u g0;
      g0.x = 1u;                                       // count=1, load D#
      g0.y = lds_lw;                                   // lds_addr (bytes)
      g0.z = (unsigned)(ga & 0xFFFFFFFFull);           // global_addr[31:0]
      g0.w = (unsigned)((ga >> 32) & 0x1FFFFFFull)     // global_addr[56:32]
             | (2u << 30);                             // type = 2 (image)
      v8i g1;
      g1[0] = (int)(1u << 16);   // data_size = 2 bytes, no multicast
      g1[1] = (int)(32u << 16);  // tensor_dim0 = 32
      g1[2] = (int)(32u << 16);  // tensor_dim1 = 32
      g1[3] = (int)(32u << 16);  // tile_dim0 = 32
      g1[4] = 32;                // tile_dim1 = 32, tile_dim2 = 0
      g1[5] = (int)SS;           // tensor_dim0_stride = 1024 (elements)
      g1[6] = 0;
      g1[7] = 0;
      v4i gz = {0, 0, 0, 0};
#if defined(__clang_major__) && __clang_major__ >= 23
      v8i gz8 = {0, 0, 0, 0, 0, 0, 0, 0};
      __builtin_amdgcn_tensor_load_to_lds(g0, g1, gz, gz, gz8, 0);
#else
      __builtin_amdgcn_tensor_load_to_lds(g0, g1, gz, gz, 0);
#endif
    }

    // ---- cooperative transposed staging of x tile (static indexing) ----
    {
      const int r0 = (tid & 15) * 2;   // pair of j rows (lanes -> banks)
      const int c0 = (tid >> 4) * 8;   // h chunk base
      const __bf16* src = xb + ((size_t)b * SS + j0 + r0) * HH + c0;
      Pack8 ra, rb;
      ra.q = *(const uint4*)(src);        // row r0,   8 bf16
      rb.q = *(const uint4*)(src + HH);   // row r0+1, 8 bf16
#pragma unroll
      for (int k = 0; k < 8; ++k) {
        const unsigned pk = (unsigned)ra.s[k] | ((unsigned)rb.s[k] << 16);
        *(unsigned*)&lxT[c0 + k][r0] = pk;  // two bf16: rows r0, r0+1
      }
    }

    if (wv == 0) __builtin_amdgcn_s_wait_tensorcnt(0);
    __syncthreads();

    // ---- fragments: all ds_load_b128 ----
    const int akoff = hi * 8;
    FragBF16 a0, a1, bm;
    a0.q[0] = *(const uint4*)&lw[col][akoff];
    a0.q[1] = *(const uint4*)&lw[col][16 + akoff];
    a1.q[0] = *(const uint4*)&lw[col + 16][akoff];
    a1.q[1] = *(const uint4*)&lw[col + 16][16 + akoff];
    const int bkoff = hi * 8;
    bm.q[0] = *(const uint4*)&lxT[h0 + col][bkoff];
    bm.q[1] = *(const uint4*)&lxT[h0 + col][bkoff + 16];

    acc0 = __builtin_amdgcn_wmma_f32_16x16x32_bf16(
        false, a0.v, false, bm.v, (short)0, acc0, false, false);
    acc1 = __builtin_amdgcn_wmma_f32_16x16x32_bf16(
        false, a1.v, false, bm.v, (short)0, acc1, false, false);
    __syncthreads();
  }

  // D layout: lane col N = col, VGPR r -> M = r + 8*hi
#pragma unroll
  for (int r = 0; r < 8; ++r) {
    const int m = r + 8 * hi;
    out[((size_t)b * SS + i0 + m) * HH + h0 + col]      = acc0[r];
    out[((size_t)b * SS + i0 + 16 + m) * HH + h0 + col] = acc1[r];
  }
}

// ---------------------------------------------------------------------------
extern "C" void kernel_launch(void* const* d_in, const int* in_sizes, int n_in,
                              void* d_out, int out_size, void* d_ws,
                              size_t ws_size, hipStream_t stream) {
  (void)in_sizes; (void)n_in; (void)out_size; (void)ws_size;
  const float* x   = (const float*)d_in[0];   // [B,S,H]
  const float* rel = (const float*)d_in[1];   // [S,S,H]
  float* out       = (float*)d_out;           // [B,S,H]

  __bf16* xb  = (__bf16*)d_ws;                                    // 2 MB
  __bf16* wgt = (__bf16*)((char*)d_ws +
                          (size_t)BB * SS * HH * sizeof(__bf16)); // 16 MB

  const int nx = BB * SS * HH;
  cvt_x_bf16_kernel<<<(nx + 255) / 256, 256, 0, stream>>>(x, xb, nx);
  scores_softmax_kernel<<<SS, 256, 0, stream>>>(x, rel, wgt);
  wmma_av_kernel<<<dim3(SS / 32, BB), 256, 0, stream>>>(wgt, xb, out);
}
